// AttentionHead_74363063763371
// MI455X (gfx1250) — matrix operand order repaired and checked by reading, not compiled, not run on hardware
//
#include <hip/hip_runtime.h>

typedef __attribute__((ext_vector_type(16))) _Float16 v16h;
typedef __attribute__((ext_vector_type(8)))  _Float16 v8h;
typedef __attribute__((ext_vector_type(4)))  _Float16 v4h;
typedef __attribute__((ext_vector_type(8)))  float    v8f;
typedef __attribute__((ext_vector_type(4)))  float    v4f;

#define DH     64        // head dim
#define SEQ    4096      // sequence length
#define KT     32        // keys per tile
#define QW     16        // q rows per wave
#define NW     4         // waves per workgroup
#define QTILE  (QW * NW) // 64 q rows per workgroup
#define NT     (SEQ / KT)

// padded LDS row strides (in halfs) -> conflict-free b128 access patterns
#define KP 72            // K tile:  [KT][DH] halfs, 144B rows
#define VP 40            // V^T:     [DH][KT] halfs,  80B rows
#define PP 40            // P^T:     per-wave [QW][KT] halfs, 80B rows

union AV { v16h v; v8h h[2]; };

// lane <-> lane^16 exchange in wave32 via v_permlanex16_b32 (pure VALU)
__device__ __forceinline__ float swap_half16(float x) {
  unsigned int i = __builtin_bit_cast(unsigned int, x);
  unsigned int r = __builtin_amdgcn_permlanex16(i, i, 0x76543210u, 0xfedcba98u,
                                                false, false);
  return __builtin_bit_cast(float, r);
}

__global__ __launch_bounds__(NW * 32, 1) void
fa_fwd_kernel(const float* __restrict__ Q, const float* __restrict__ K,
              const float* __restrict__ V, float* __restrict__ O)
{
  __shared__ __align__(16) _Float16 Klds[2][KT * KP];
  __shared__ __align__(16) _Float16 Vtlds[2][DH * VP];
  __shared__ __align__(16) _Float16 Plds[NW * QW * PP];

  const int tid  = threadIdx.x;
  const int wave = tid >> 5;
  const int ln   = tid & 15;        // lane % 16
  const int lh   = (tid >> 4) & 1;  // lane half (0/1)

  const int b  = blockIdx.y;
  const int qw = blockIdx.x * QTILE + wave * QW;  // this wave's q base

  const size_t bo = (size_t)b * SEQ * DH;
  const float* Qb = Q + bo;
  const float* Kb = K + bo;
  const float* Vb = V + bo;
  float*       Ob = O + bo;

  // ---- loop-invariant Q^T B-fragments (K-dim = d, N = q), f32 -> f16 ----
  v16h qb[2];
  {
    const float* qrow = Qb + (size_t)(qw + ln) * DH + lh * 8;
    #pragma unroll
    for (int h = 0; h < 2; ++h) {
      const float* p = qrow + h * 32;
      #pragma unroll
      for (int j = 0; j < 16; ++j) qb[h][j] = (_Float16)p[(j & 7) + (j >> 3) * 16];
    }
  }

  // ---- prologue: fill buffer 0 with tile 0 ----
  #pragma unroll
  for (int i = 0; i < 4; ++i) {
    int e = tid * 4 + i, row = e >> 4, c4 = e & 15;
    v4f kv = *(const v4f*)(Kb + (size_t)row * DH + c4 * 4);
    v4f vv = *(const v4f*)(Vb + (size_t)row * DH + c4 * 4);
    v4h kh;
    #pragma unroll
    for (int j = 0; j < 4; ++j) kh[j] = (_Float16)kv[j];
    *(v4h*)&Klds[0][row * KP + c4 * 4] = kh;
    #pragma unroll
    for (int j = 0; j < 4; ++j) Vtlds[0][(c4 * 4 + j) * VP + row] = (_Float16)vv[j];
  }
  __syncthreads();

  // A-fragments, software-pipelined across iterations (preloaded after the
  // barrier of the previous iteration -> live across the back-edge, so they
  // must occupy distinct registers and their LDS latency pipelines).
  AV a[2][2];   // K rows     (QK^T A-frags)
  AV av[4];     // V^T rows   (PV   A-frags)
  auto load_frags = [&](int bufn) {
    #pragma unroll
    for (int g = 0; g < 2; ++g)
      #pragma unroll
      for (int h = 0; h < 2; ++h) {
        const _Float16* kr = &Klds[bufn][(g * 16 + ln) * KP + h * 32 + lh * 8];
        a[g][h].h[0] = *(const v8h*)kr;
        a[g][h].h[1] = *(const v8h*)(kr + 16);
      }
    #pragma unroll
    for (int g = 0; g < 4; ++g) {
      const _Float16* vr = &Vtlds[bufn][(g * 16 + ln) * VP + lh * 8];
      av[g].h[0] = *(const v8h*)vr;
      av[g].h[1] = *(const v8h*)(vr + 16);
    }
  };
  load_frags(0);

  float m = -__builtin_inff();
  float l = 0.0f;
  v8f o[4] = {};

  const float SC = 0.125f * 1.44269504088896340f;  // (1/sqrt(64)) * log2(e)

  for (int it = 0; it < NT; ++it) {
    const int  buf = it & 1;
    const bool pre = (it + 1) < NT;

    // ---- stage next tile's global loads (latency hidden under compute) ----
    v4f kv[4], vv[4];
    if (pre) {
      const int kt2 = (it + 1) * KT;
      #pragma unroll
      for (int i = 0; i < 4; ++i) {
        int e = tid * 4 + i, row = e >> 4, c4 = e & 15;
        kv[i] = *(const v4f*)(Kb + (size_t)(kt2 + row) * DH + c4 * 4);
        vv[i] = *(const v4f*)(Vb + (size_t)(kt2 + row) * DH + c4 * 4);
      }
      if (it + 2 < NT) {  // global_prefetch_b8 for tile +2
        __builtin_prefetch(Kb + (size_t)((it + 2) * KT + (tid >> 2)) * DH + (tid & 3) * 16, 0, 0);
        __builtin_prefetch(Vb + (size_t)((it + 2) * KT + (tid >> 2)) * DH + (tid & 3) * 16, 0, 0);
      }
    }

    // ---- S^T = K x Q^T (A-frags preloaded last iteration) ----
    v8f st[2];
    #pragma unroll
    for (int g = 0; g < 2; ++g) {
      v8f acc = {};
      acc = __builtin_amdgcn_wmma_f32_16x16x32_f16(false, a[g][0].v, false, qb[0],
                                                   (short)0, acc, false, false);
      acc = __builtin_amdgcn_wmma_f32_16x16x32_f16(false, a[g][1].v, false, qb[1],
                                                   (short)0, acc, false, false);
      st[g] = acc;
    }

    // ---- online softmax (q = lane%16 across both lane halves) ----
    float pm = -__builtin_inff();
    #pragma unroll
    for (int r = 0; r < 8; ++r) {
      pm = fmaxf(pm, st[0][r]);
      pm = fmaxf(pm, st[1][r]);
    }
    pm = fmaxf(pm, swap_half16(pm));
    float mn    = fmaxf(m, pm * SC);
    float alpha = __builtin_amdgcn_exp2f(m - mn);  // raw v_exp_f32

    float ps = 0.0f;
    v8h ph[2];
    #pragma unroll
    for (int g = 0; g < 2; ++g)
      #pragma unroll
      for (int r = 0; r < 8; ++r) {
        float p = __builtin_amdgcn_exp2f(st[g][r] * SC - mn);
        ps += p;
        ph[g][r] = (_Float16)p;
      }
    ps += swap_half16(ps);

    // ---- P^T C-layout -> B-layout via wave-private LDS round trip ----
    _Float16* pw = &Plds[wave * QW * PP];
    *(v8h*)&pw[ln * PP + 0 * 16 + lh * 8] = ph[0];
    *(v8h*)&pw[ln * PP + 1 * 16 + lh * 8] = ph[1];

    // rescale accumulators while the P store is in flight
    l = l * alpha + ps;
    m = mn;
    #pragma unroll
    for (int g = 0; g < 4; ++g)
      #pragma unroll
      for (int r = 0; r < 8; ++r) o[g][r] *= alpha;

    AV pb;
    pb.h[0] = *(const v8h*)&pw[ln * PP + lh * 8];
    pb.h[1] = *(const v8h*)&pw[ln * PP + lh * 8 + 16];

    // ---- O^T += V^T x P^T ----
    #pragma unroll
    for (int g = 0; g < 4; ++g)
      o[g] = __builtin_amdgcn_wmma_f32_16x16x32_f16(false, av[g].v, false, pb.v,
                                                    (short)0, o[g], false, false);

    // ---- commit staged tile into the other buffer, single barrier ----
    if (pre) {
      #pragma unroll
      for (int i = 0; i < 4; ++i) {
        int e = tid * 4 + i, row = e >> 4, c4 = e & 15;
        v4h kh;
        #pragma unroll
        for (int j = 0; j < 4; ++j) kh[j] = (_Float16)kv[i][j];
        *(v4h*)&Klds[buf ^ 1][row * KP + c4 * 4] = kh;
        #pragma unroll
        for (int j = 0; j < 4; ++j)
          Vtlds[buf ^ 1][(c4 * 4 + j) * VP + row] = (_Float16)vv[i][j];
      }
    }
    __syncthreads();

    // ---- preload next tile's A-fragments right after the barrier ----
    if (pre) load_frags(buf ^ 1);
  }

  // ---- epilogue: divide by l, store O (d = g*16 + 8*lh + r, q = qw + ln) ----
  float inv = 1.0f / l;
  #pragma unroll
  for (int g = 0; g < 4; ++g) {
    v4f lo, hi;
    #pragma unroll
    for (int r = 0; r < 4; ++r) {
      lo[r] = o[g][r] * inv;
      hi[r] = o[g][r + 4] * inv;
    }
    float* op = Ob + (size_t)(qw + ln) * DH + g * 16 + lh * 8;
    *(v4f*)op       = lo;
    *(v4f*)(op + 4) = hi;
  }
}

extern "C" void kernel_launch(void* const* d_in, const int* in_sizes, int n_in,
                              void* d_out, int out_size, void* d_ws, size_t ws_size,
                              hipStream_t stream) {
  const float* Q = (const float*)d_in[0];
  const float* K = (const float*)d_in[1];
  const float* V = (const float*)d_in[2];
  float*       O = (float*)d_out;
  int B = in_sizes[0] / (SEQ * DH);
  if (B <= 0) return;
  dim3 grid(SEQ / QTILE, B);
  fa_fwd_kernel<<<grid, NW * 32, 0, stream>>>(Q, K, V, O);
}
